// MetaUpscale_15857019257115
// MI455X (gfx1250) — matrix operand order repaired and checked by reading, not compiled, not run on hardware
//
#include <hip/hip_runtime.h>
#include <hip/hip_bf16.h>

typedef __attribute__((ext_vector_type(16))) _Float16 v16h;
typedef __attribute__((ext_vector_type(8)))  _Float16 v8h;
typedef __attribute__((ext_vector_type(8)))  float    v8f;

#define IN_C   64
#define HH     96
#define WW     96
#define OW     192
#define KRED   576            // IN_C * 3 * 3
#define WSTR   1728           // KRED * OUT_C
#define TILE_W 8              // input pixels (waves) per block
#define WIN_ELEMS (4 * IN_C * 3 * (TILE_W + 2))   // 7680 source window elems
#define PK_ELEMS  (TILE_W * 4 * KRED)             // 18432 f16 = 36 KB

// One block = 8 waves over 8 adjacent input pixels (one row), all 4 batches.
// Each wave computes the 2x2 output block of its input pixel:
//   A (16x32 f16) : rows = batches (rows 4..15 duplicate row m&3; harmless, rows
//                   of D are independent and we only store rows 0..3)
//   B (32x16 f16) : cols = (dy,dx,o) meta-weights (cols 12..15 duplicate 0..3)
//   D (16x16 f32) : K=576 accumulated via 18x v_wmma_f32_16x16x32_f16
//
// Patch data is staged in LDS pre-permuted into reduction order pk[wave][b][k]
// (k = c*9+ky*3+kx), so the A fragment per chunk is two aligned ds_load_b128
// with NO per-element address math in the hot loop.
__global__ __launch_bounds__(256)
void meta_upscale_wmma(const float* __restrict__ x,
                       const float* __restrict__ weight,
                       float* __restrict__ out) {
    __shared__ _Float16 pk[PK_ELEMS];   // [wave][b][k], 16B-aligned per (wave,b)

    const int tid = threadIdx.x;
    const int by  = blockIdx.x / (WW / TILE_W);             // input row
    const int bx  = (blockIdx.x % (WW / TILE_W)) * TILE_W;  // first input col

    // ---- Stage: read each window element once, scatter to its <=3 (wave,kx) slots ----
    for (int e = tid; e < WIN_ELEMS; e += 256) {
        const int cc = e % 10;           // window col (gx = bx + cc - 1)
        const int r  = (e / 10) % 3;     // ky
        const int bc = e / 30;           // b*64 + c
        const int b  = bc >> 6;
        const int c  = bc & 63;
        const int gy = by + r - 1;
        const int gx = bx + cc - 1;
        float v = 0.0f;
        if ((unsigned)gy < (unsigned)HH && (unsigned)gx < (unsigned)WW)
            v = x[(bc * HH + gy) * WW + gx];
        const _Float16 h = (_Float16)v;
        const int kb = c * 9 + r * 3;
        #pragma unroll
        for (int kx = 0; kx < 3; ++kx) {
            const int wv = cc - kx;                 // wave that uses this (kx) tap
            if ((unsigned)wv < (unsigned)TILE_W)
                pk[(wv * 4 + b) * KRED + kb + kx] = h;
        }
    }
    __syncthreads();

    const int wave = tid >> 5;
    const int lane = tid & 31;
    const int m    = lane & 15;
    const int hi   = lane >> 4;
    const int xq   = bx + wave;

    // A source: v8h view of this wave's packed patch, row = batch (dup for m>=4)
    const v8h* pkv = reinterpret_cast<const v8h*>(pk);
    const int aBase = (wave * 4 + (m & 3)) * (KRED / 8) + hi;   // +kc*4 per chunk

    // B column decode; lanes 12..15 duplicate lanes 0..3 (same addresses -> coalesced)
    const int mm = (m < 12) ? m : (m - 12);
    const int dy = mm / 6;
    const int dx = (mm % 6) / 3;
    const int o  = mm % 3;
    const long p = (long)(2 * by + dy) * OW + (2 * xq + dx);
    const float* wp = weight + (size_t)p * WSTR + o + (size_t)hi * 24;  // hi*8 k-offset

    v8f acc = {};

    #pragma unroll 3
    for (int kc = 0; kc < KRED / 32; ++kc) {
        // ---- A: two aligned 16B LDS loads (K = hi*8+t and K = 16+hi*8+t runs) ----
        union { v16h v; v8h h[2]; } a;
        a.h[0] = pkv[aBase + kc * 4];
        a.h[1] = pkv[aBase + kc * 4 + 2];

        // ---- B: 16 nontemporal stride-3 loads, immediate offsets ----
        const float* wq = wp + (size_t)kc * 96;   // kbase*3
        v16h bm;
        #pragma unroll
        for (int t = 0; t < 16; ++t)
            bm[t] = (_Float16)__builtin_nontemporal_load(wq + t * 3 + (t >= 8 ? 24 : 0));

        __builtin_prefetch(wq + 192, 0, 1);       // 2 chunks ahead in the stream

        acc = __builtin_amdgcn_wmma_f32_16x16x32_f16(
            /*neg_a=*/false, a.v, /*neg_b=*/false, bm,
            /*c_mod=*/(short)0, acc, /*reuse_a=*/false, /*reuse_b=*/false);
    }

    // ---- D: VGPR r -> M = r (hi==0 half), lane -> N. Store the real 4x12 tile. ----
    if (hi == 0 && m < 12) {
        const int i = 2 * by + dy;
        const int j = 2 * xq + dx;
        #pragma unroll
        for (int r = 0; r < 4; ++r)   // batches
            out[((r * 3 + o) * OW + i) * OW + j] = acc[r];
    }
}

extern "C" void kernel_launch(void* const* d_in, const int* in_sizes, int n_in,
                              void* d_out, int out_size, void* d_ws, size_t ws_size,
                              hipStream_t stream) {
    (void)in_sizes; (void)n_in; (void)d_ws; (void)ws_size; (void)out_size;
    const float* x = (const float*)d_in[0];
    const float* w = (const float*)d_in[1];
    float* out     = (float*)d_out;

    dim3 block(256);                       // 8 wave32 waves
    dim3 grid(HH * (WW / TILE_W));         // 96 * 12 = 1152 blocks
    hipLaunchKernelGGL(meta_upscale_wmma, grid, block, 0, stream, x, w, out);
}
